// TransformerEncoderLayer_22634477650048
// MI455X (gfx1250) — matrix operand order repaired and checked by reading, not compiled, not run on hardware
//
#include <hip/hip_runtime.h>

typedef _Float16 f16;
typedef __attribute__((ext_vector_type(16))) _Float16 v16h;
typedef __attribute__((ext_vector_type(8)))  float    v8f;
typedef __attribute__((ext_vector_type(4)))  unsigned int u32x4;
typedef __attribute__((ext_vector_type(8)))  int          i32x8;
typedef __attribute__((ext_vector_type(4)))  int          i32x4;

#define B_   4
#define S_   2048
#define D_   1024
#define H_   16
#define HD_  64
#define DFF_ 4096
#define NROW (B_ * S_)   // 8192 rows

// ---------------------------------------------------------------------------
// Tensor Data Mover availability (ROCm7.2: 5-arg builtin; therock: 6-arg)
// ---------------------------------------------------------------------------
#if defined(__has_builtin)
#if __has_builtin(__builtin_amdgcn_tensor_load_to_lds)
#define HAVE_TDM 1
#endif
#endif
#ifndef HAVE_TDM
#define HAVE_TDM 0
#endif
#if defined(__has_include)
#if __has_include(<hip/amd_detail/amd_gfx1250_TDM.h>)
#define TDM_6ARG 1
#endif
#endif
#ifndef TDM_6ARG
#define TDM_6ARG 0
#endif

#if HAVE_TDM
// 2D tile load: tile_rows rows of tile_k f16 elements, global row stride
// row_stride elems, packed contiguously into LDS at lds_off.
// D# layout per CDNA5 ISA.
static __device__ inline void tdm_load_2d(unsigned lds_off, const void* gptr,
                                          unsigned tile_k, unsigned tile_rows,
                                          unsigned long long row_stride) {
  unsigned long long ga = (unsigned long long)(uintptr_t)gptr;
  unsigned td0 = (unsigned)row_stride;   // tensor_dim0 (full row in-bounds)
  unsigned td1 = tile_rows;              // tensor_dim1
  u32x4 g0 = {
      1u,                                           // count=1, user mode
      lds_off,                                      // lds_addr (bytes)
      (unsigned)(ga & 0xffffffffu),                 // global_addr[31:0]
      (unsigned)((ga >> 32) & 0x01ffffffu) | (2u << 30)  // addr[56:32]|type=2
  };
  i32x8 g1 = {
      (int)(1u << 16),                                             // data_size=2B
      (int)((td0 & 0xffffu) << 16),                                // dim0[15:0]
      (int)(((td0 >> 16) & 0xffffu) | ((td1 & 0xffffu) << 16)),    // dim0_hi|dim1_lo
      (int)(((td1 >> 16) & 0xffffu) | ((tile_k & 0xffffu) << 16)), // dim1_hi|tile0
      (int)(tile_rows & 0xffffu),                                  // tile1 (tile2=0)
      (int)(unsigned)(row_stride & 0xffffffffu),                   // stride0 lo
      (int)(unsigned)((row_stride >> 32) & 0xffffu),               // stride0 hi
      0
  };
  i32x4 z = {0, 0, 0, 0};
#if TDM_6ARG
  i32x8 z8 = {0, 0, 0, 0, 0, 0, 0, 0};
  __builtin_amdgcn_tensor_load_to_lds(g0, g1, z, z, z8, 0);
#else
  __builtin_amdgcn_tensor_load_to_lds(g0, g1, z, z, 0);
#endif
}
#endif

static __device__ inline unsigned lds_off_of(const void* p) {
  // LDS aperture: flat addr[31:0] == LDS byte address (ISA §10.2)
  return (unsigned)((uintptr_t)p & 0xffffffffu);
}

// ---------------------------------------------------------------------------
// WMMA helper: wave32 16x16x32 f16 -> f32 accumulate
// ---------------------------------------------------------------------------
union FragU { v16h v; uint4 q[2]; };
union AccU  { v8f  v; float f[8]; };

static __device__ inline v8f wmma_f16(v16h a, v16h b, v8f c) {
  return __builtin_amdgcn_wmma_f32_16x16x32_f16(false, a, false, b, (short)0, c,
                                                false, false);
}

// ---------------------------------------------------------------------------
// fp32 -> f16 conversion (vectorized, n multiple of 4)
// ---------------------------------------------------------------------------
__global__ void cvt_f32_f16(const float* __restrict__ in, f16* __restrict__ out,
                            int n4) {
  int i = blockIdx.x * blockDim.x + threadIdx.x;
  int stride = gridDim.x * blockDim.x;
  for (; i < n4; i += stride) {
    float4 v = ((const float4*)in)[i];
    union { f16 h[4]; uint2 u; } o;
    o.h[0] = (f16)v.x; o.h[1] = (f16)v.y; o.h[2] = (f16)v.z; o.h[3] = (f16)v.w;
    ((uint2*)out)[i] = o.u;
  }
}

// fp32 [K,N] -> f16 transposed [N,K] (tiled through LDS)
__global__ __launch_bounds__(256) void cvt_t_f16(const float* __restrict__ in,
                                                 f16* __restrict__ out,
                                                 int K, int N) {
  __shared__ float tile[16][17];
  int k0 = blockIdx.y * 16, n0 = blockIdx.x * 16;
  int tx = threadIdx.x, ty = threadIdx.y;
  tile[ty][tx] = in[(size_t)(k0 + ty) * N + n0 + tx];
  __syncthreads();
  out[(size_t)(n0 + ty) * K + k0 + tx] = (f16)tile[tx][ty];
}

// ---------------------------------------------------------------------------
// GEMM: C[M,N] = A[M,K](f16) * Bt[N,K](f16, pre-transposed) + bias.
// Block = 128x128 tile, 256 threads (8 waves). Wave (wm=w>>1, wn=w&1) owns a
// 32x64 sub-tile: 2 m-tiles x 4 n-tiles of 16x16 (8 fp32 accumulators).
// K staged 64 at a time, double-buffered via TDM (s_wait_tensorcnt 2 keeps
// one stage of DMA in flight behind compute).
// OUT_MODE: 0 = f16 row-major, 1 = f32 row-major, 2 = f16 transposed [N,M].
// ---------------------------------------------------------------------------
template <bool RELU, int OUT_MODE>
__global__ __launch_bounds__(256) void gemm_f16_kernel(
    const f16* __restrict__ A, const f16* __restrict__ Bt,
    const float* __restrict__ bias, void* __restrict__ out,
    int M, int N, int K) {
  __shared__ __align__(16) f16 As[2][128 * 64];   // [m][k]
  __shared__ __align__(16) f16 Bs[2][128 * 64];   // [n][k]

  const int tid  = threadIdx.x;
  const int wave = tid >> 5;
  const int lane = tid & 31;
  const int ln   = lane & 15;
  const int kh   = lane >> 4;
  const int m0   = blockIdx.y * 128;
  const int n0   = blockIdx.x * 128;
  const int wm   = wave >> 1;      // 0..3 -> m offset wm*32
  const int wn   = wave & 1;       // 0..1 -> n offset wn*64

  v8f acc[2][4];
#pragma unroll
  for (int mt = 0; mt < 2; ++mt)
#pragma unroll
    for (int nt = 0; nt < 4; ++nt) acc[mt][nt] = (v8f){};

  const int nstage = K >> 6;

#if HAVE_TDM
  if (wave == 0) {  // prologue: stage 0 in flight
    tdm_load_2d(lds_off_of(As[0]), &A[(size_t)m0 * K], 64, 128,
                (unsigned long long)K);
    tdm_load_2d(lds_off_of(Bs[0]), &Bt[(size_t)n0 * K], 64, 128,
                (unsigned long long)K);
  }
#endif

  for (int s = 0; s < nstage; ++s) {
    const int cur = s & 1;
    const int k0  = s << 6;
#if HAVE_TDM
    if (wave == 0) {
      if (s + 1 < nstage) {  // issue next stage into alternate buffer
        tdm_load_2d(lds_off_of(As[cur ^ 1]), &A[(size_t)m0 * K + k0 + 64], 64,
                    128, (unsigned long long)K);
        tdm_load_2d(lds_off_of(Bs[cur ^ 1]), &Bt[(size_t)n0 * K + k0 + 64], 64,
                    128, (unsigned long long)K);
        __builtin_amdgcn_s_wait_tensorcnt(2);  // stage s complete
      } else {
        __builtin_amdgcn_s_wait_tensorcnt(0);
      }
    }
#else
    {
      int row = tid >> 1, ch = (tid & 1) * 32;
#pragma unroll
      for (int c = 0; c < 4; ++c)
        *(uint4*)&As[cur][row * 64 + ch + c * 8] =
            *(const uint4*)&A[(size_t)(m0 + row) * K + k0 + ch + c * 8];
#pragma unroll
      for (int c = 0; c < 4; ++c)
        *(uint4*)&Bs[cur][row * 64 + ch + c * 8] =
            *(const uint4*)&Bt[(size_t)(n0 + row) * K + k0 + ch + c * 8];
    }
#endif
    __syncthreads();

#pragma unroll
    for (int ks = 0; ks < 2; ++ks) {
      FragU a0, a1;
      {
        const f16* p = &As[cur][(wm * 32 + ln) * 64 + ks * 32];
        a0.q[0] = *(const uint4*)(p + 8 * kh);
        a0.q[1] = *(const uint4*)(p + 16 + 8 * kh);
        p += 16 * 64;
        a1.q[0] = *(const uint4*)(p + 8 * kh);
        a1.q[1] = *(const uint4*)(p + 16 + 8 * kh);
      }
#pragma unroll
      for (int nt = 0; nt < 4; ++nt) {
        FragU b;
        const f16* p =
            &Bs[cur][(wn * 64 + nt * 16 + ln) * 64 + ks * 32 + 8 * kh];
        b.q[0] = *(const uint4*)p;
        b.q[1] = *(const uint4*)(p + 16);
        acc[0][nt] = wmma_f16(a0.v, b.v, acc[0][nt]);
        acc[1][nt] = wmma_f16(a1.v, b.v, acc[1][nt]);
      }
    }
    __syncthreads();
  }

#pragma unroll
  for (int mt = 0; mt < 2; ++mt) {
#pragma unroll
    for (int nt = 0; nt < 4; ++nt) {
      AccU ac; ac.v = acc[mt][nt];
      int col = n0 + wn * 64 + nt * 16 + ln;
      float bs = bias[col];
#pragma unroll
      for (int r = 0; r < 8; ++r) {
        int row = m0 + wm * 32 + mt * 16 + r + 8 * kh;
        float vv = ac.f[r] + bs;
        if (RELU) vv = vv > 0.f ? vv : 0.f;
        if (OUT_MODE == 0)      ((f16*)out)[(size_t)row * N + col] = (f16)vv;
        else if (OUT_MODE == 1) ((float*)out)[(size_t)row * N + col] = vv;
        else                    ((f16*)out)[(size_t)col * M + row] = (f16)vv;
      }
    }
  }
}

// ---------------------------------------------------------------------------
// Flash attention: grid (S/64, H, B), 128 threads (4 waves).
// Q/K row-major [NROW,D]; V pre-transposed [D,NROW]. Wave owns 16 q rows.
// K/V tiles double-buffered via TDM.
// ---------------------------------------------------------------------------
__global__ __launch_bounds__(128) void attention_kernel(
    const f16* __restrict__ Q, const f16* __restrict__ Km,
    const f16* __restrict__ Vt, f16* __restrict__ O) {
  __shared__ __align__(16) f16 Ks[2][64 * 64];    // [key][dim]
  __shared__ __align__(16) f16 Vs[2][64 * 64];    // [dim][key]
  __shared__ __align__(16) f16 Ps[4][16 * 64];    // per-wave P [row][key]

  const int tid  = threadIdx.x;
  const int wave = tid >> 5;
  const int lane = tid & 31;
  const int ln   = lane & 15;
  const int kh   = lane >> 4;
  const int h    = blockIdx.y;
  const int b    = blockIdx.z;
  const size_t base  = ((size_t)b * S_) * D_ + (size_t)h * HD_;
  const size_t vbase = (size_t)h * HD_ * NROW + (size_t)b * S_;
  const int q0 = blockIdx.x * 64 + wave * 16;

  FragU qf[2];
  {
    const f16* qrow = Q + base + (size_t)(q0 + ln) * D_;
#pragma unroll
    for (int ks = 0; ks < 2; ++ks) {
      qf[ks].q[0] = *(const uint4*)&qrow[ks * 32 + 8 * kh];
      qf[ks].q[1] = *(const uint4*)&qrow[ks * 32 + 16 + 8 * kh];
    }
  }

  float mrun[8], lrun[8];
  v8f oacc[4];
#pragma unroll
  for (int r = 0; r < 8; ++r) { mrun[r] = -1e30f; lrun[r] = 0.f; }
#pragma unroll
  for (int nt = 0; nt < 4; ++nt) oacc[nt] = (v8f){};

#if HAVE_TDM
  if (wave == 0) {  // prologue: first K/V tile in flight
    tdm_load_2d(lds_off_of(Ks[0]), Km + base, 64, 64, (unsigned long long)D_);
    tdm_load_2d(lds_off_of(Vs[0]), Vt + vbase, 64, 64,
                (unsigned long long)NROW);
  }
#endif

  for (int sblk = 0; sblk < (S_ / 64); ++sblk) {
    const int cur = sblk & 1;
    const int kb  = sblk * 64;
#if HAVE_TDM
    if (wave == 0) {
      if (kb + 64 < S_) {
        tdm_load_2d(lds_off_of(Ks[cur ^ 1]), Km + base + (size_t)(kb + 64) * D_,
                    64, 64, (unsigned long long)D_);
        tdm_load_2d(lds_off_of(Vs[cur ^ 1]), Vt + vbase + kb + 64, 64, 64,
                    (unsigned long long)NROW);
        __builtin_amdgcn_s_wait_tensorcnt(2);
      } else {
        __builtin_amdgcn_s_wait_tensorcnt(0);
      }
    }
#else
    {
      int row = tid >> 1, ch = (tid & 1) * 32;
      const f16* kr = Km + base + (size_t)(kb + row) * D_;
#pragma unroll
      for (int c = 0; c < 4; ++c)
        *(uint4*)&Ks[cur][row * 64 + ch + c * 8] =
            *(const uint4*)&kr[ch + c * 8];
      const f16* vr = Vt + vbase + (size_t)row * NROW + kb;  // row = dim here
#pragma unroll
      for (int c = 0; c < 4; ++c)
        *(uint4*)&Vs[cur][row * 64 + ch + c * 8] =
            *(const uint4*)&vr[ch + c * 8];
    }
#endif
    __syncthreads();

    // scores: 4 tiles of 16x16 over 64 keys
    AccU sc[4];
#pragma unroll
    for (int nt = 0; nt < 4; ++nt) {
      v8f a = {};
#pragma unroll
      for (int ks = 0; ks < 2; ++ks) {
        FragU bf;
        const f16* kp = &Ks[cur][(nt * 16 + ln) * 64 + ks * 32 + 8 * kh];
        bf.q[0] = *(const uint4*)kp; bf.q[1] = *(const uint4*)(kp + 16);
        a = wmma_f16(qf[ks].v, bf.v, a);
      }
      sc[nt].v = a;
    }

    // online softmax (rows live in 16-lane halves)
    float mnew[8];
#pragma unroll
    for (int r = 0; r < 8; ++r) {
      float mx = -1e30f;
#pragma unroll
      for (int nt = 0; nt < 4; ++nt) {
        sc[nt].f[r] *= 0.125f;  // 1/sqrt(HD)
        mx = fmaxf(mx, sc[nt].f[r]);
      }
#pragma unroll
      for (int d = 1; d < 16; d <<= 1) mx = fmaxf(mx, __shfl_xor(mx, d, 32));
      mnew[r] = fmaxf(mrun[r], mx);
    }
#pragma unroll
    for (int r = 0; r < 8; ++r) {
      float rs = 0.f;
#pragma unroll
      for (int nt = 0; nt < 4; ++nt) {
        float e = __expf(sc[nt].f[r] - mnew[r]);
        sc[nt].f[r] = e;
        rs += e;
      }
#pragma unroll
      for (int d = 1; d < 16; d <<= 1) rs += __shfl_xor(rs, d, 32);
      float scale = __expf(mrun[r] - mnew[r]);
      lrun[r] = lrun[r] * scale + rs;
      mrun[r] = mnew[r];
#pragma unroll
      for (int nt = 0; nt < 4; ++nt) oacc[nt][r] *= scale;
    }

    // C-layout -> A-layout for P via wave-private LDS tile
#pragma unroll
    for (int nt = 0; nt < 4; ++nt)
#pragma unroll
      for (int r = 0; r < 8; ++r)
        Ps[wave][(r + 8 * kh) * 64 + nt * 16 + ln] = (f16)sc[nt].f[r];

    FragU pf[2];
#pragma unroll
    for (int ks = 0; ks < 2; ++ks) {
      const f16* pp = &Ps[wave][ln * 64 + ks * 32 + 8 * kh];
      pf[ks].q[0] = *(const uint4*)pp;
      pf[ks].q[1] = *(const uint4*)(pp + 16);
    }
#pragma unroll
    for (int nt = 0; nt < 4; ++nt) {
#pragma unroll
      for (int ks = 0; ks < 2; ++ks) {
        FragU bf;
        const f16* vp = &Vs[cur][(nt * 16 + ln) * 64 + ks * 32 + 8 * kh];
        bf.q[0] = *(const uint4*)vp; bf.q[1] = *(const uint4*)(vp + 16);
        oacc[nt] = wmma_f16(pf[ks].v, bf.v, oacc[nt]);
      }
    }
    __syncthreads();
  }

#pragma unroll
  for (int nt = 0; nt < 4; ++nt)
#pragma unroll
    for (int r = 0; r < 8; ++r) {
      float ov = oacc[nt][r] / lrun[r];
      O[base + (size_t)(q0 + r + 8 * kh) * D_ + nt * 16 + ln] = (f16)ov;
    }
}

// ---------------------------------------------------------------------------
// Residual add + LayerNorm: one block (256 thr) per row of D_=1024.
// ---------------------------------------------------------------------------
static __device__ inline float block_reduce_sum(float v, float* red) {
#pragma unroll
  for (int d = 1; d < 32; d <<= 1) v += __shfl_xor(v, d, 32);
  int wave = threadIdx.x >> 5, lane = threadIdx.x & 31;
  if (lane == 0) red[wave] = v;
  __syncthreads();
  float s = 0.f;
#pragma unroll
  for (int i = 0; i < 8; ++i) s += red[i];
  __syncthreads();
  return s;
}

template <bool WRITE_F16>
__global__ __launch_bounds__(256) void add_ln_kernel(
    const float* __restrict__ x, const float* __restrict__ y,
    const float* __restrict__ g, const float* __restrict__ be,
    float* __restrict__ out_f32, f16* __restrict__ out_f16) {
  __shared__ float red[8];
  const size_t row = blockIdx.x;
  const float* xr = x + row * D_;
  const float* yr = y + row * D_;
  float v[4];
  float s = 0.f;
#pragma unroll
  for (int i = 0; i < 4; ++i) {
    int c = threadIdx.x + i * 256;
    v[i] = xr[c] + yr[c];
    s += v[i];
  }
  s = block_reduce_sum(s, red);
  float mu = s * (1.f / D_);
  float s2 = 0.f;
#pragma unroll
  for (int i = 0; i < 4; ++i) { float d = v[i] - mu; s2 += d * d; }
  s2 = block_reduce_sum(s2, red);
  float rstd = rsqrtf(s2 * (1.f / D_) + 1e-6f);
#pragma unroll
  for (int i = 0; i < 4; ++i) {
    int c = threadIdx.x + i * 256;
    float o = (v[i] - mu) * rstd * g[c] + be[c];
    out_f32[row * D_ + c] = o;
    if (WRITE_F16) out_f16[row * D_ + c] = (f16)o;
  }
}

// ---------------------------------------------------------------------------
// Orchestration
// ---------------------------------------------------------------------------
static void run_cvt(const float* in, f16* out, int n, hipStream_t s) {
  int n4 = n / 4;
  int blocks = (n4 + 255) / 256;
  if (blocks > 4096) blocks = 4096;
  cvt_f32_f16<<<blocks, 256, 0, s>>>(in, out, n4);
}

static void run_cvt_t(const float* in, f16* out, int K, int N, hipStream_t s) {
  cvt_t_f16<<<dim3(N / 16, K / 16), dim3(16, 16), 0, s>>>(in, out, K, N);
}

extern "C" void kernel_launch(void* const* d_in, const int* in_sizes, int n_in,
                              void* d_out, int out_size, void* d_ws,
                              size_t ws_size, hipStream_t stream) {
  const float* x   = (const float*)d_in[0];
  const float* Wq  = (const float*)d_in[1];
  const float* bq  = (const float*)d_in[2];
  const float* Wk  = (const float*)d_in[3];
  const float* bk  = (const float*)d_in[4];
  const float* Wv  = (const float*)d_in[5];
  const float* bv  = (const float*)d_in[6];
  const float* Wo  = (const float*)d_in[7];
  const float* bo  = (const float*)d_in[8];
  const float* W1  = (const float*)d_in[9];
  const float* b1  = (const float*)d_in[10];
  const float* W2  = (const float*)d_in[11];
  const float* b2  = (const float*)d_in[12];
  const float* g1  = (const float*)d_in[13];
  const float* be1 = (const float*)d_in[14];
  const float* g2  = (const float*)d_in[15];
  const float* be2 = (const float*)d_in[16];

  char* ws = (char*)d_ws;
  size_t off = 0;
  auto alloc = [&](size_t bytes) -> void* {
    void* p = ws + off;
    off = (off + bytes + 255) & ~(size_t)255;
    return p;
  };
  f16*   xb    = (f16*)alloc((size_t)NROW * D_ * 2);
  f16*   Wqt   = (f16*)alloc((size_t)D_ * D_ * 2);     // [N][K]
  f16*   Wkt   = (f16*)alloc((size_t)D_ * D_ * 2);
  f16*   Wvt   = (f16*)alloc((size_t)D_ * D_ * 2);
  f16*   Wot   = (f16*)alloc((size_t)D_ * D_ * 2);
  f16*   W1t   = (f16*)alloc((size_t)D_ * DFF_ * 2);   // [DFF][D]
  f16*   W2t   = (f16*)alloc((size_t)DFF_ * D_ * 2);   // [D][DFF]
  f16*   Qb    = (f16*)alloc((size_t)NROW * D_ * 2);
  f16*   Kb    = (f16*)alloc((size_t)NROW * D_ * 2);
  f16*   Vtb   = (f16*)alloc((size_t)NROW * D_ * 2);   // [D][NROW]
  f16*   attnb = (f16*)alloc((size_t)NROW * D_ * 2);
  float* projf = (float*)alloc((size_t)NROW * D_ * 4);
  float* x1f   = (float*)alloc((size_t)NROW * D_ * 4);
  f16*   x1b   = (f16*)alloc((size_t)NROW * D_ * 2);
  f16*   ffn1b = (f16*)alloc((size_t)NROW * DFF_ * 2);
  float* ffn2f = (float*)alloc((size_t)NROW * D_ * 4);

  // 1) conversions: x -> f16; weights -> f16 transposed [N][K]
  run_cvt(x, xb, NROW * D_, stream);
  run_cvt_t(Wq, Wqt, D_, D_, stream);
  run_cvt_t(Wk, Wkt, D_, D_, stream);
  run_cvt_t(Wv, Wvt, D_, D_, stream);
  run_cvt_t(Wo, Wot, D_, D_, stream);
  run_cvt_t(W1, W1t, D_, DFF_, stream);
  run_cvt_t(W2, W2t, DFF_, D_, stream);

  // 2) Q, K row-major; V written transposed [D][NROW]
  dim3 gQKV(D_ / 128, NROW / 128);
  gemm_f16_kernel<false, 0><<<gQKV, 256, 0, stream>>>(xb, Wqt, bq, Qb, NROW, D_, D_);
  gemm_f16_kernel<false, 0><<<gQKV, 256, 0, stream>>>(xb, Wkt, bk, Kb, NROW, D_, D_);
  gemm_f16_kernel<false, 2><<<gQKV, 256, 0, stream>>>(xb, Wvt, bv, Vtb, NROW, D_, D_);

  // 3) attention
  dim3 gAttn(S_ / 64, H_, B_);
  attention_kernel<<<gAttn, 128, 0, stream>>>(Qb, Kb, Vtb, attnb);

  // 4) O projection (fp32 out for residual)
  gemm_f16_kernel<false, 1><<<gQKV, 256, 0, stream>>>(attnb, Wot, bo, projf, NROW, D_, D_);

  // 5) x1 = LN(x + proj)
  add_ln_kernel<true><<<NROW, 256, 0, stream>>>(x, projf, g1, be1, x1f, x1b);

  // 6) FFN1 with ReLU (f16 out)
  dim3 gF1(DFF_ / 128, NROW / 128);
  gemm_f16_kernel<true, 0><<<gF1, 256, 0, stream>>>(x1b, W1t, b1, ffn1b, NROW, DFF_, D_);

  // 7) FFN2 (fp32 out)
  dim3 gF2(D_ / 128, NROW / 128);
  gemm_f16_kernel<false, 1><<<gF2, 256, 0, stream>>>(ffn1b, W2t, b2, ffn2f, NROW, D_, DFF_);

  // 8) out = LN(x1 + ffn2)
  add_ln_kernel<false><<<NROW, 256, 0, stream>>>(x1f, ffn2f, g2, be2, (float*)d_out, nullptr);

  (void)in_sizes; (void)n_in; (void)out_size; (void)ws_size;
}
